// NpuFlatQuantMLP_43207370998075
// MI455X (gfx1250) — 8-bit matrix operand order repaired and checked by reading, compiled, not run on hardware
//
#include <hip/hip_runtime.h>
#include <stdint.h>


typedef __attribute__((ext_vector_type(8))) int v8i;

#define SEQ   1024
#define H_DIM 4096
#define I_DIM 14336
#define L1D   64
#define R1D   64
#define L2D   112
#define R2D   128

__device__ __forceinline__ int nib_perm(int k) {
  return ((k & 1) << 2) | (k & 8) | ((k >> 1) & 3);
}

__device__ __forceinline__ v8i load_a_frag(const uint8_t* p) {
  const int* q = (const int*)p;
  v8i a;
  a[0] = q[0];  a[1] = q[1];
  a[2] = q[4];  a[3] = q[5];
  a[4] = q[8];  a[5] = q[9];
  a[6] = q[12]; a[7] = q[13];
  return a;
}

__device__ __forceinline__ v8i load_b_frag_packed(const uint8_t* p) {
  uint32_t w0 = *(const uint32_t*)(p + 0);
  uint32_t w1 = *(const uint32_t*)(p + 8);
  uint32_t w2 = *(const uint32_t*)(p + 16);
  uint32_t w3 = *(const uint32_t*)(p + 24);
  v8i b;
  b[0] = (int)(((w0      ) & 0x0F0F0F0Fu) ^ 0x08080808u);
  b[1] = (int)(((w0 >> 4 ) & 0x0F0F0F0Fu) ^ 0x08080808u);
  b[2] = (int)(((w1      ) & 0x0F0F0F0Fu) ^ 0x08080808u);
  b[3] = (int)(((w1 >> 4 ) & 0x0F0F0F0Fu) ^ 0x08080808u);
  b[4] = (int)(((w2      ) & 0x0F0F0F0Fu) ^ 0x08080808u);
  b[5] = (int)(((w2 >> 4 ) & 0x0F0F0F0Fu) ^ 0x08080808u);
  b[6] = (int)(((w3      ) & 0x0F0F0F0Fu) ^ 0x08080808u);
  b[7] = (int)(((w3 >> 4 ) & 0x0F0F0F0Fu) ^ 0x08080808u);
  return b;
}

__global__ __launch_bounds__(256) void k1_transform_quant_ug(
    const float* __restrict__ x, const float* __restrict__ left,
    const float* __restrict__ right, int8_t* __restrict__ xq,
    float* __restrict__ a_scale, float* __restrict__ rsum) {
  __shared__ float lX[H_DIM];
  __shared__ float lY[H_DIM];
  __shared__ float red[256];
  __shared__ int   redi[256];
  const int t = blockIdx.x;
  const int tid = threadIdx.x;
  const float* xrow = x + (size_t)t * H_DIM;
  for (int i = tid; i < H_DIM; i += 256) lX[i] = xrow[i];
  __syncthreads();
  for (int i = tid; i < H_DIM; i += 256) {
    const int l = i >> 6, n = i & 63;
    float acc = 0.f;
#pragma unroll 8
    for (int r = 0; r < R1D; ++r) acc += lX[l * 64 + r] * right[r * 64 + n];
    lY[i] = acc;
  }
  __syncthreads();
  float z[16];
  float amax = 0.f;
#pragma unroll
  for (int ii = 0; ii < 16; ++ii) {
    const int i = ii * 256 + tid;
    const int m = i >> 6, n = i & 63;
    float acc = 0.f;
#pragma unroll 8
    for (int l = 0; l < L1D; ++l) acc += left[m * 64 + l] * lY[l * 64 + n];
    z[ii] = acc;
    amax = fmaxf(amax, fabsf(acc));
  }
  red[tid] = amax;
  __syncthreads();
  for (int s = 128; s > 0; s >>= 1) {
    if (tid < s) red[tid] = fmaxf(red[tid], red[tid + s]);
    __syncthreads();
  }
  const float scale = red[0] * (1.0f / 7.0f);
  if (tid == 0) a_scale[t] = scale;
  const float inv = 1.0f / scale;
  int8_t* qrow = xq + (size_t)t * H_DIM;
  int rs = 0;
#pragma unroll
  for (int ii = 0; ii < 16; ++ii) {
    const int i = ii * 256 + tid;
    float q = rintf(z[ii] * inv);
    q = fminf(fmaxf(q, -8.f), 7.f);
    const int qi = (int)q;
    rs += qi;
    qrow[(i & ~15) | nib_perm(i & 15)] = (int8_t)qi;
  }
  redi[tid] = rs;
  __syncthreads();
  for (int s = 128; s > 0; s >>= 1) {
    if (tid < s) redi[tid] += redi[tid + s];
    __syncthreads();
  }
  if (tid == 0) rsum[t] = (float)redi[0];
}

__global__ __launch_bounds__(256) void k2_gemm_upgate(
    const int8_t* __restrict__ xq, const float* __restrict__ a_scale,
    const float* __restrict__ rsum,
    const uint8_t* __restrict__ wU, const uint8_t* __restrict__ wG,
    const float* __restrict__ sU, const float* __restrict__ sG,
    float* __restrict__ hbuf) {
  const int tid  = threadIdx.x;
  const int lane = tid & 31;
  const int wave = tid >> 5;
  const int lo16 = lane & 15;
  const int hi   = lane >> 4;
  const int mBase = blockIdx.y * 64 + (wave & 3) * 16;
  const int nBase = blockIdx.x * 128 + (wave >> 2) * 64;

  v8i accU[4], accG[4];
#pragma unroll
  for (int j = 0; j < 4; ++j)
#pragma unroll
    for (int e = 0; e < 8; ++e) { accU[j][e] = 0; accG[j][e] = 0; }

  const uint8_t* aRow =
      (const uint8_t*)xq + (size_t)(mBase + lo16) * H_DIM + hi * 8;

#pragma unroll 2
  for (int kb = 0; kb < H_DIM; kb += 64) {
    const v8i A = load_a_frag(aRow + kb);
#pragma unroll
    for (int j = 0; j < 4; ++j) {
      const int o = nBase + j * 16 + lo16;
      const size_t po = (size_t)o * (H_DIM / 2) + (kb >> 1) + hi * 4;
      __builtin_prefetch(wU + po + 512, 0, 3);
      const v8i BU = load_b_frag_packed(wU + po);
      accU[j] = __builtin_amdgcn_wmma_i32_16x16x64_iu8(
          true, A, false, BU, accU[j], false, false);
      const v8i BG = load_b_frag_packed(wG + po);
      accG[j] = __builtin_amdgcn_wmma_i32_16x16x64_iu8(
          true, A, false, BG, accG[j], false, false);
    }
  }

#pragma unroll
  for (int j = 0; j < 4; ++j) {
    const int n = nBase + j * 16 + lo16;
    const float su = sU[n];
    const float sg = sG[n];
#pragma unroll
    for (int e = 0; e < 8; ++e) {
      const int m = mBase + hi * 8 + e;
      const float sa = a_scale[m];
      const float c8 = 8.0f * rsum[m];
      const float u = ((float)accU[j][e] - c8) * sa * su;
      const float g = ((float)accG[j][e] - c8) * sa * sg;
      const float hv = (g / (1.0f + __expf(-g))) * u;
      hbuf[(size_t)m * I_DIM + n] = hv;
    }
  }
}

__global__ __launch_bounds__(128) void k3a_right_inplace(
    float* __restrict__ hbuf, const float* __restrict__ right) {
  __shared__ float sH[R2D];
  const int l = blockIdx.x;
  const int t = blockIdx.y;
  const int n = threadIdx.x;
  float* row = hbuf + (size_t)t * I_DIM + l * R2D;
  sH[n] = row[n];
  __syncthreads();
  float acc = 0.f;
#pragma unroll 8
  for (int r = 0; r < R2D; ++r) acc += sH[r] * right[r * R2D + n];
  row[n] = acc;
}

__global__ __launch_bounds__(256) void k3b_left_quant(
    const float* __restrict__ hbuf, const float* __restrict__ left,
    int8_t* __restrict__ hq, float* __restrict__ a_scale,
    float* __restrict__ rsum) {
  __shared__ float sY[I_DIM];
  __shared__ float red[256];
  __shared__ int   redi[256];
  const int t = blockIdx.x;
  const int tid = threadIdx.x;
  const float* yrow = hbuf + (size_t)t * I_DIM;
  for (int i = tid; i < I_DIM; i += 256) sY[i] = yrow[i];
  __syncthreads();
  float z[56];
  float amax = 0.f;
  for (int ii = 0; ii < 56; ++ii) {
    const int i = ii * 256 + tid;
    const int m = i >> 7;
    const int n = i & 127;
    float acc = 0.f;
#pragma unroll 8
    for (int l = 0; l < L2D; ++l) acc += left[m * L2D + l] * sY[l * R2D + n];
    z[ii] = acc;
    amax = fmaxf(amax, fabsf(acc));
  }
  red[tid] = amax;
  __syncthreads();
  for (int s = 128; s > 0; s >>= 1) {
    if (tid < s) red[tid] = fmaxf(red[tid], red[tid + s]);
    __syncthreads();
  }
  const float scale = red[0] * (1.0f / 7.0f);
  if (tid == 0) a_scale[t] = scale;
  const float inv = 1.0f / scale;
  int8_t* qrow = hq + (size_t)t * I_DIM;
  int rs = 0;
  for (int ii = 0; ii < 56; ++ii) {
    const int i = ii * 256 + tid;
    float q = rintf(z[ii] * inv);
    q = fminf(fmaxf(q, -8.f), 7.f);
    const int qi = (int)q;
    rs += qi;
    qrow[(i & ~15) | nib_perm(i & 15)] = (int8_t)qi;
  }
  redi[tid] = rs;
  __syncthreads();
  for (int s = 128; s > 0; s >>= 1) {
    if (tid < s) redi[tid] += redi[tid + s];
    __syncthreads();
  }
  if (tid == 0) rsum[t] = (float)redi[0];
}

__global__ __launch_bounds__(256) void k4_gemm_down(
    const int8_t* __restrict__ hq, const float* __restrict__ a_scale,
    const float* __restrict__ rsum,
    const uint8_t* __restrict__ wD, const float* __restrict__ sD,
    float* __restrict__ out) {
  const int tid  = threadIdx.x;
  const int lane = tid & 31;
  const int wave = tid >> 5;
  const int lo16 = lane & 15;
  const int hi   = lane >> 4;
  const int mBase = blockIdx.y * 64 + (wave & 3) * 16;
  const int nBase = blockIdx.x * 128 + (wave >> 2) * 64;

  v8i acc[4];
#pragma unroll
  for (int j = 0; j < 4; ++j)
#pragma unroll
    for (int e = 0; e < 8; ++e) acc[j][e] = 0;

  const uint8_t* aRow =
      (const uint8_t*)hq + (size_t)(mBase + lo16) * I_DIM + hi * 8;

#pragma unroll 2
  for (int kb = 0; kb < I_DIM; kb += 64) {
    const v8i A = load_a_frag(aRow + kb);
#pragma unroll
    for (int j = 0; j < 4; ++j) {
      const int o = nBase + j * 16 + lo16;
      const size_t po = (size_t)o * (I_DIM / 2) + (kb >> 1) + hi * 4;
      __builtin_prefetch(wD + po + 512, 0, 3);
      const v8i B = load_b_frag_packed(wD + po);
      acc[j] = __builtin_amdgcn_wmma_i32_16x16x64_iu8(
          true, A, false, B, acc[j], false, false);
    }
  }

#pragma unroll
  for (int j = 0; j < 4; ++j) {
    const int n = nBase + j * 16 + lo16;
    const float sd = sD[n];
#pragma unroll
    for (int e = 0; e < 8; ++e) {
      const int m = mBase + hi * 8 + e;
      const float c8 = 8.0f * rsum[m];
      out[(size_t)m * H_DIM + n] = ((float)acc[j][e] - c8) * a_scale[m] * sd;
    }
  }
}

extern "C" void kernel_launch(void* const* d_in, const int* in_sizes, int n_in,
                              void* d_out, int out_size, void* d_ws,
                              size_t ws_size, hipStream_t stream) {
  (void)in_sizes; (void)n_in; (void)out_size; (void)ws_size;
  const float*   x        = (const float*)d_in[0];
  const float*   ug_left  = (const float*)d_in[1];
  const float*   ug_right = (const float*)d_in[2];
  const float*   dn_left  = (const float*)d_in[3];
  const float*   dn_right = (const float*)d_in[4];
  const float*   s_up     = (const float*)d_in[5];
  const float*   s_gate   = (const float*)d_in[6];
  const float*   s_down   = (const float*)d_in[7];
  const uint8_t* w_up     = (const uint8_t*)d_in[8];
  const uint8_t* w_gate   = (const uint8_t*)d_in[9];
  const uint8_t* w_down   = (const uint8_t*)d_in[10];

  uint8_t* ws = (uint8_t*)d_ws;
  size_t off = 0;
  int8_t* xq    = (int8_t*)(ws + off); off += (size_t)SEQ * H_DIM;
  float*  a_ug  = (float*)(ws + off);  off += (size_t)SEQ * sizeof(float);
  float*  rs_ug = (float*)(ws + off);  off += (size_t)SEQ * sizeof(float);
  float*  hbuf  = (float*)(ws + off);  off += (size_t)SEQ * I_DIM * 4;
  int8_t* hq    = (int8_t*)(ws + off); off += (size_t)SEQ * I_DIM;
  float*  a_dn  = (float*)(ws + off);  off += (size_t)SEQ * sizeof(float);
  float*  rs_dn = (float*)(ws + off);  off += (size_t)SEQ * sizeof(float);

  k1_transform_quant_ug<<<SEQ, 256, 0, stream>>>(x, ug_left, ug_right, xq,
                                                 a_ug, rs_ug);
  k2_gemm_upgate<<<dim3(I_DIM / 128, SEQ / 64), 256, 0, stream>>>(
      xq, a_ug, rs_ug, w_up, w_gate, s_up, s_gate, hbuf);
  k3a_right_inplace<<<dim3(L2D, SEQ), 128, 0, stream>>>(hbuf, dn_right);
  k3b_left_quant<<<SEQ, 256, 0, stream>>>(hbuf, dn_left, hq, a_dn, rs_dn);
  k4_gemm_down<<<dim3(H_DIM / 128, SEQ / 64), 256, 0, stream>>>(
      hq, a_dn, rs_dn, w_down, s_down, (float*)d_out);
}
